// ARRBM_19112604467253
// MI455X (gfx1250) — matrix operand order repaired and checked by reading, not compiled, not run on hardware
//
#include <hip/hip_runtime.h>

typedef __attribute__((ext_vector_type(16))) _Float16 v16h;
typedef __attribute__((ext_vector_type(8)))  _Float16 v8h;
typedef __attribute__((ext_vector_type(8)))  float    v8f;

#define NV     128
#define NH     256
#define WD     4160
#define BATCHN 16384
#define NSTEP  64
#define WROW   136   // 128 + 8 halfs pad (16B-aligned rows, staggers LDS banks)

// Transposed formulation: per 16x16 WMMA tile, A = W tile (M=hidden, K) from LDS,
// B = X^T tile (K x batch) from registers.  C-layout => lane owns one batch row,
// 8 hidden values per tile => hidden-product reduction is ONE shfl_xor(16) per c.
template<int KC>
__device__ __forceinline__ void arrbm_step_body(
    const float* __restrict__ x, float* __restrict__ ws,
    const _Float16* wlds, const float* blds,
    const float* c0ld, const float* c1ld,
    const int i, const int j)
{
    const int tid  = threadIdx.x;
    const int wave = tid >> 5;
    const int lane = tid & 31;
    const int lr   = lane & 15;      // batch col within tile (B/C N index)
    const int hiK  = lane >> 4;

    const int tilesTotal = BATCHN >> 4;       // 1024
    const int stride     = gridDim.x * 8;

    for (int t = blockIdx.x * 8 + wave; t < tilesTotal; t += stride) {
        const int r0 = t << 4;
        const float* xr = x + (r0 + lr) * NV;   // this lane's batch row

        // ---- B = X^T: lane holds batch col lr; slots t -> K = kc*32 + 16*hiK + t
        v16h b[KC == 0 ? 1 : KC];
        #pragma unroll
        for (int kc = 0; kc < KC; ++kc) {
            const float* base = xr + (kc << 5) + (hiK << 4);
            float4 f0 = *(const float4*)(base);
            float4 f1 = *(const float4*)(base + 4);
            float4 f2 = *(const float4*)(base + 8);
            float4 f3 = *(const float4*)(base + 12);
            v16h bv;
            bv[0]=(_Float16)f0.x;  bv[1]=(_Float16)f0.y;  bv[2]=(_Float16)f0.z;  bv[3]=(_Float16)f0.w;
            bv[4]=(_Float16)f1.x;  bv[5]=(_Float16)f1.y;  bv[6]=(_Float16)f1.z;  bv[7]=(_Float16)f1.w;
            bv[8]=(_Float16)f2.x;  bv[9]=(_Float16)f2.y;  bv[10]=(_Float16)f2.z; bv[11]=(_Float16)f2.w;
            bv[12]=(_Float16)f3.x; bv[13]=(_Float16)f3.y; bv[14]=(_Float16)f3.z; bv[15]=(_Float16)f3.w;
            b[kc] = bv;
        }

        float p0 = 1.0f, p1 = 1.0f, p2 = 1.0f, p3 = 1.0f;

        #pragma unroll 2
        for (int n = 0; n < 16; ++n) {
            // ---- A = W tile from LDS: lane M-row = lr (hidden), two 16B runs/chunk
            const _Float16* ap = wlds + ((n << 4) + lr) * WROW + (hiK << 4);
            v8f acc = {0.f, 0.f, 0.f, 0.f, 0.f, 0.f, 0.f, 0.f};
            #pragma unroll
            for (int kc = 0; kc < KC; ++kc) {
                v8h a0v = *(const v8h*)(ap + (kc << 5));
                v8h a1v = *(const v8h*)(ap + (kc << 5) + 8);
                v16h av;
                #pragma unroll
                for (int tt = 0; tt < 8; ++tt) { av[tt] = a0v[tt]; av[8 + tt] = a1v[tt]; }
                acc = __builtin_amdgcn_wmma_f32_16x16x32_f16(
                          false, av, false, b[kc], (short)0, acc, false, false);
            }
            // acc[r] -> hidden h = n*16 + 8*hiK + r ; bias / W_cur as 8-float runs
            const int hb8 = (n << 4) + (hiK << 3);
            float4 bA = *(const float4*)(blds + hb8);
            float4 bB = *(const float4*)(blds + hb8 + 4);
            float4 w0A = *(const float4*)(c0ld + hb8);
            float4 w0B = *(const float4*)(c0ld + hb8 + 4);
            float4 w1A = *(const float4*)(c1ld + hb8);
            float4 w1B = *(const float4*)(c1ld + hb8 + 4);
            float bia[8] = {bA.x, bA.y, bA.z, bA.w, bB.x, bB.y, bB.z, bB.w};
            float wc0[8] = {w0A.x, w0A.y, w0A.z, w0A.w, w0B.x, w0B.y, w0B.z, w0B.w};
            float wc1[8] = {w1A.x, w1A.y, w1A.z, w1A.w, w1B.x, w1B.y, w1B.z, w1B.w};
            #pragma unroll
            for (int r = 0; r < 8; ++r) {
                float a0 = acc[r] + bia[r];
                float s01 = wc0[r] + wc1[r];
                p0 *= __cosf(a0);
                p1 *= __cosf(a0 + wc0[r]);
                p2 *= __cosf(a0 + wc1[r]);
                p3 *= __cosf(a0 + s01);
            }
        }

        // ---- combine the two hidden halves: one butterfly step per c
        p0 *= __shfl_xor(p0, 16, 32);
        p1 *= __shfl_xor(p1, 16, 32);
        p2 *= __shfl_xor(p2, 16, 32);
        p3 *= __shfl_xor(p3, 16, 32);

        // ---- lanes 0..15 each write their own batch row
        if (lane < 16) {
            const int row = r0 + lr;
            float nrm = __builtin_sqrtf(p0*p0 + p1*p1 + p2*p2 + p3*p3);
            nrm = fmaxf(nrm, 1e-14f);
            float v0 = xr[j];
            float v1 = xr[j + 1];
            int idx = (int)(v0 + 2.0f * v1);
            float sel = (idx == 0) ? p0 : (idx == 1) ? p1 : (idx == 2) ? p2 : p3;
            ws[i * BATCHN + row] = sel / nrm;
        }
    }
}

__global__ __launch_bounds__(256)
void ARRBM_step_kernel(const float* __restrict__ x,
                       const float* __restrict__ w,
                       const float* __restrict__ hb,
                       float* __restrict__ ws) {
    extern __shared__ char smem[];
    _Float16* wlds = (_Float16*)smem;                       // NH * WROW halfs
    float*    blds = (float*)(smem + NH * WROW * 2);        // NH bias
    float*    c0ld = blds + NH;
    float*    c1ld = c0ld + NH;

    const int i  = blockIdx.y;
    const int j  = 2 * i;
    const int s  = i * (i + 1);
    const int KC = (j + 31) >> 5;
    const int tid = threadIdx.x;

    // stage step weights (f32 -> f16, zero-padded to KC*32)
    for (int kc = 0; kc < KC; ++kc) {
        for (int e = tid; e < NH * 32; e += 256) {
            int h = e >> 5;
            int k = (kc << 5) + (e & 31);
            float v = (k < j) ? w[h * WD + s + k] : 0.0f;
            wlds[h * WROW + k] = (_Float16)v;
        }
    }
    {
        int h = tid;                 // block is exactly 256 threads
        blds[h] = hb[h];
        c0ld[h] = w[h * WD + s + j];
        c1ld[h] = w[h * WD + s + j + 1];
    }
    __syncthreads();

    switch (KC) {
        case 0:  arrbm_step_body<0>(x, ws, wlds, blds, c0ld, c1ld, i, j); break;
        case 1:  arrbm_step_body<1>(x, ws, wlds, blds, c0ld, c1ld, i, j); break;
        case 2:  arrbm_step_body<2>(x, ws, wlds, blds, c0ld, c1ld, i, j); break;
        case 3:  arrbm_step_body<3>(x, ws, wlds, blds, c0ld, c1ld, i, j); break;
        default: arrbm_step_body<4>(x, ws, wlds, blds, c0ld, c1ld, i, j); break;
    }
}

__global__ __launch_bounds__(256)
void ARRBM_combine_kernel(const float* __restrict__ ws, float* __restrict__ out) {
    int n = blockIdx.x * blockDim.x + threadIdx.x;
    if (n < BATCHN) {
        float p = 1.0f;
        #pragma unroll 8
        for (int st = 0; st < NSTEP; ++st)
            p *= ws[st * BATCHN + n];
        out[n] = p;
    }
}

extern "C" void kernel_launch(void* const* d_in, const int* in_sizes, int n_in,
                              void* d_out, int out_size, void* d_ws, size_t ws_size,
                              hipStream_t stream) {
    (void)in_sizes; (void)n_in; (void)out_size; (void)ws_size;
    const float* x  = (const float*)d_in[0];   // (16384, 128)
    const float* w  = (const float*)d_in[1];   // (256, 4160)
    const float* hb = (const float*)d_in[2];   // (256,)
    float* out = (float*)d_out;                // (16384,)
    float* ws  = (float*)d_ws;                 // 64 * 16384 f32 = 4 MB

    const size_t shmem = (size_t)NH * WROW * sizeof(_Float16) + 3 * NH * sizeof(float);
    dim3 grid(16, NSTEP, 1);
    ARRBM_step_kernel<<<grid, 256, shmem, stream>>>(x, w, hb, ws);
    ARRBM_combine_kernel<<<(BATCHN + 255) / 256, 256, 0, stream>>>(ws, out);
}
